// GNNLayer_88648124989935
// MI455X (gfx1250) — matrix operand order repaired and checked by reading, not compiled, not run on hardware
//
#include <hip/hip_runtime.h>
#include <hip/hip_bf16.h>
#include <math.h>

// ---------------------------------------------------------------------------
// GNN layer for MI455X (gfx1250, wave32, WMMA + TDM).
//   N=8192 nodes, DIM=128.
//   Dominant op: agg = (A @ msg)/deg  -> bf16 WMMA. A streamed from HBM
//   (256MB, the roofline), msg tile staged in LDS by the Tensor Data Mover
//   (double buffered, TENSORcnt-tracked). GRU matmuls also bf16 WMMA.
// ---------------------------------------------------------------------------

#define NNODES 8192
#define DIM    128

typedef __attribute__((ext_vector_type(16))) __bf16       v16bf;
typedef __attribute__((ext_vector_type(8)))  float        v8f;
typedef __attribute__((ext_vector_type(4)))  unsigned int u32x4;
typedef __attribute__((ext_vector_type(4)))  int          i32x4;
typedef __attribute__((ext_vector_type(8)))  int          i32x8;

union BF16x16 {                 // all members trivially copyable
  u32x4        q[2];
  unsigned int u[8];
  v16bf        v;
};

static __device__ __forceinline__ v8f wmma_bf16(v16bf a, v16bf b, v8f c) {
  // D = A(16x32) * B(32x16) + C, f32 accumulate
  return __builtin_amdgcn_wmma_f32_16x16x32_bf16(
      /*neg_a=*/false, a, /*neg_b=*/false, b,
      /*c_mod=*/(short)0, c, /*reuse_a=*/false, /*reuse_b=*/false);
}

static __device__ __forceinline__ float sigmoidf_(float x) {
  return 1.0f / (1.0f + __expf(-x));
}

// ---------------------------------------------------------------------------
// Tensor Data Mover support (gfx1250). Builtin arity differs across the two
// known toolchains; discriminate via the therock-only TDM header.
// ---------------------------------------------------------------------------
#if defined(__has_builtin)
#if __has_builtin(__builtin_amdgcn_tensor_load_to_lds)
#define USE_TDM 1
#endif
#endif

#if defined(__has_include)
#if __has_include(<hip/amd_detail/amd_gfx1250_TDM.h>)
#define TDM_SIX_ARGS 1
#endif
#endif

#ifdef USE_TDM
// Issue a TDM DMA of one contiguous 32x128 bf16 tile (8KB) of msgb[tile kt]
// into LDS at byte address lds_byte_addr. D# built per CDNA5 ISA ch.8:
//   group0: count=1 | lds_addr | global_addr(57b) | type=2
//   group1: data_size=2B, tensor_dim0=N*DIM, tensor_dim1=1,
//           tile_dim0=4096 elems, tile_dim1=1, stride0=N*DIM
static __device__ __forceinline__ void tdm_issue(const __bf16* msgb, int kt,
                                                 unsigned lds_byte_addr) {
  unsigned long long ga =
      (unsigned long long)(const void*)(msgb + (size_t)kt * 32 * DIM);
  u32x4 g0;
  g0[0] = 1u;                                              // count=1
  g0[1] = lds_byte_addr;                                   // lds_addr
  g0[2] = (unsigned)ga;                                    // global_addr[31:0]
  g0[3] = (unsigned)((ga >> 32) & 0x01FFFFFFull) | (2u << 30);  // [56:32]|type=2
  i32x8 g1;
  g1[0] = 0x00010000;   // workgroup_mask=0, data_size=1 (2 bytes)
  g1[1] = 0;            // atomic_barrier_addr=0, tensor_dim0[15:0]=0 (2^20)
  g1[2] = 0x00010010;   // tensor_dim0[31:16]=0x10, tensor_dim1=1
  g1[3] = 0x10000000;   // tile_dim0 = 4096 elements
  g1[4] = 1;            // tile_dim1 = 1, tile_dim2 = 0
  g1[5] = 0x00100000;   // tensor_dim0_stride = N*DIM
  g1[6] = 0;
  g1[7] = 0;
  i32x4 z4 = {};
#ifdef TDM_SIX_ARGS
  i32x8 z8 = {};
  __builtin_amdgcn_tensor_load_to_lds(g0, g1, z4, z4, z8, 0);
#else
  __builtin_amdgcn_tensor_load_to_lds(g0, g1, z4, z4, 0);
#endif
}
#endif  // USE_TDM

// ---------------------------------------------------------------------------
// Kernel 0: convert+transpose GRU weights to bf16:  wT[k*384 + j] = w[j*128 + k]
// ---------------------------------------------------------------------------
__global__ void k_prep(const float* __restrict__ w_ih, const float* __restrict__ w_hh,
                       __bf16* __restrict__ wihT, __bf16* __restrict__ whhT) {
  int idx = blockIdx.x * blockDim.x + threadIdx.x;   // 0..49151
  int j = idx / DIM;                                  // 0..383
  int k = idx - j * DIM;                              // 0..127
  wihT[k * 384 + j] = (__bf16)w_ih[idx];
  whhT[k * 384 + j] = (__bf16)w_hh[idx];
}

// ---------------------------------------------------------------------------
// Kernel 1: messages = relu(X @ Wmsg^T + b)  (fp32 math, bf16 output)
//           also emits xb = bf16(X) for the GRU WMMA path.
// ---------------------------------------------------------------------------
__global__ void __launch_bounds__(256) k_msg(
    const float* __restrict__ X, const float* __restrict__ Wm,
    const float* __restrict__ bm,
    __bf16* __restrict__ msgb, __bf16* __restrict__ xb) {
  __shared__ float xs[16 * DIM];
  const int t  = threadIdx.x;
  const int nb = blockIdx.x * 16;
#pragma unroll
  for (int i = 0; i < 8; ++i) {
    int idx = t + i * 256;
    float v = X[(size_t)nb * DIM + idx];
    xs[idx] = v;
    xb[(size_t)nb * DIM + idx] = (__bf16)v;
  }
  __syncthreads();
  const int c = t & 127;       // output column
  const int p = t >> 7;        // node half: nodes p*8 .. p*8+7
  float acc[8] = {};
  for (int k = 0; k < DIM; ++k) {
    float wv = Wm[c * DIM + k];
#pragma unroll
    for (int j = 0; j < 8; ++j)
      acc[j] = fmaf(xs[(p * 8 + j) * DIM + k], wv, acc[j]);
  }
  float bias = bm[c];
#pragma unroll
  for (int j = 0; j < 8; ++j) {
    float m = acc[j] + bias;
    m = m > 0.0f ? m : 0.0f;
    msgb[(size_t)(nb + p * 8 + j) * DIM + c] = (__bf16)m;
  }
}

// ---------------------------------------------------------------------------
// Kernel 2: agg = (A @ msg) / clamp(rowsum(A),1)   -> bf16 output
// 128 threads = 4 waves; wave owns 16 output rows x 128 cols (8 C tiles).
// msg tile (32x128 bf16 = 8KB) double-buffered in LDS via TDM (wave 0 issues,
// TENSORcnt-tracked; one workgroup barrier per K-step). A streamed with b128
// loads; degrees accumulated for free in the same pass.
// ---------------------------------------------------------------------------
__global__ void __launch_bounds__(128) k_agg(
    const int* __restrict__ A, const __bf16* __restrict__ msgb,
    __bf16* __restrict__ aggb) {
  __shared__ __bf16 lds[2][32 * DIM];   // 2 x 8KB
  const int t    = threadIdx.x;
  const int w    = t >> 5;
  const int lane = t & 31;
  const int half = lane >> 4;
  const int row  = lane & 15;
  const int r0   = blockIdx.x * 64 + w * 16;
  const int NK   = NNODES / 32;

  v8f zero = {};
  v8f acc[8];
#pragma unroll
  for (int ct = 0; ct < 8; ++ct) acc[ct] = zero;
  unsigned degpack = 0;   // lo16 / hi16 partial degree sums (max 2048 each)

  const int* arow = A + (size_t)(r0 + row) * NNODES + half * 8;

#ifdef USE_TDM
  // flat LDS pointer low 32 bits == LDS byte offset (ISA 10.2 aperture rule)
  const unsigned ldsbase = (unsigned)(unsigned long long)(const void*)&lds[0][0];
  if (w == 0) {
    tdm_issue(msgb, 0, ldsbase);
    __builtin_amdgcn_s_wait_tensorcnt(0);
  }
  __syncthreads();
#else
  {
    const u32x4* src = (const u32x4*)msgb;
    u32x4* dst = (u32x4*)&lds[0][0];
#pragma unroll
    for (int i = 0; i < 4; ++i) dst[t * 4 + i] = src[t * 4 + i];
  }
  __syncthreads();
#endif

  for (int kt = 0; kt < NK; ++kt) {
    const bool havenext = (kt + 1 < NK);
#ifdef USE_TDM
    if (w == 0 && havenext)
      tdm_issue(msgb, kt + 1, ldsbase + (unsigned)(((kt + 1) & 1) * 8192));
#else
    u32x4 stage[4];
    if (havenext) {
      const u32x4* src = (const u32x4*)(msgb + (size_t)(kt + 1) * 32 * DIM);
#pragma unroll
      for (int i = 0; i < 4; ++i) stage[i] = src[t * 4 + i];
    }
#endif
    // prefetch A stream ahead (-> global_prefetch_b8)
    if (kt + 8 < NK) __builtin_prefetch(arow + (kt + 8) * 32, 0, 0);

    // Build A operand (16x32 bf16): lane=row, K in {h8+0..7, 16+h8+0..7}.
    // 0/1 int pair -> packed bf16 dword: (lo | hi<<16) * 0x3F80 (exact).
    const int* ap = arow + kt * 32;
    i32x4 a0 = *(const i32x4*)(ap);
    i32x4 a1 = *(const i32x4*)(ap + 4);
    i32x4 a2 = *(const i32x4*)(ap + 16);
    i32x4 a3 = *(const i32x4*)(ap + 20);
    unsigned pk[8];
    pk[0] = (unsigned)a0[0] | ((unsigned)a0[1] << 16);
    pk[1] = (unsigned)a0[2] | ((unsigned)a0[3] << 16);
    pk[2] = (unsigned)a1[0] | ((unsigned)a1[1] << 16);
    pk[3] = (unsigned)a1[2] | ((unsigned)a1[3] << 16);
    pk[4] = (unsigned)a2[0] | ((unsigned)a2[1] << 16);
    pk[5] = (unsigned)a2[2] | ((unsigned)a2[3] << 16);
    pk[6] = (unsigned)a3[0] | ((unsigned)a3[1] << 16);
    pk[7] = (unsigned)a3[2] | ((unsigned)a3[3] << 16);
    BF16x16 a;
#pragma unroll
    for (int e = 0; e < 8; ++e) {
      degpack += pk[e];
      a.u[e] = pk[e] * 0x3F80u;
    }

    // 8 column tiles: B operand column N = row, K in {h8+0..7, 16+h8+0..7}
    const __bf16* bbase = &lds[kt & 1][half * 8 * DIM + row];
#pragma unroll
    for (int ct = 0; ct < 8; ++ct) {
      BF16x16 b;
#pragma unroll
      for (int s = 0; s < 16; ++s)
        b.v[s] = bbase[(s < 8 ? s : s + 8) * DIM + ct * 16];
      acc[ct] = wmma_bf16(a.v, b.v, acc[ct]);
    }

#ifdef USE_TDM
    if (w == 0 && havenext) __builtin_amdgcn_s_wait_tensorcnt(0);
    __syncthreads();
#else
    __syncthreads();
    if (havenext) {
      u32x4* dst = (u32x4*)&lds[(kt + 1) & 1][0];
#pragma unroll
      for (int i = 0; i < 4; ++i) dst[t * 4 + i] = stage[i];
    }
    __syncthreads();
#endif
  }

  // degree: split packed halves, then lane l + l^16 hold row (l&15)'s total
  int degsum = (int)(degpack & 0xFFFFu) + (int)(degpack >> 16);
  degsum += __shfl_xor(degsum, 16, 32);
  float inv = 1.0f / fmaxf((float)degsum, 1.0f);
  float invv[8];
#pragma unroll
  for (int v = 0; v < 8; ++v) invv[v] = __shfl(inv, half * 8 + v, 32);

#pragma unroll
  for (int ct = 0; ct < 8; ++ct) {
#pragma unroll
    for (int v = 0; v < 8; ++v) {
      float val = acc[ct][v] * invv[v];
      int orow = r0 + half * 8 + v;                // C layout: M = v + half*8
      aggb[(size_t)orow * DIM + ct * 16 + row] = (__bf16)val;
    }
  }
}

// ---------------------------------------------------------------------------
// Kernel 3: GRU update via WMMA.
//   gi = agg @ wih^T, gh = h @ whh^T (bf16 WMMA, f32 accum), then fp32
//   sigmoid/tanh elementwise. 128 threads = 4 waves; wave owns 16 nodes.
// ---------------------------------------------------------------------------
__global__ void __launch_bounds__(128) k_gru(
    const __bf16* __restrict__ aggb, const __bf16* __restrict__ xb,
    const float* __restrict__ X,
    const __bf16* __restrict__ wihT, const __bf16* __restrict__ whhT,
    const float* __restrict__ b_ih, const float* __restrict__ b_hh,
    float* __restrict__ out) {
  const int t     = threadIdx.x;
  const int w     = t >> 5;
  const int lane  = t & 31;
  const int half  = lane >> 4;
  const int row   = lane & 15;
  const int node0 = (blockIdx.x * 4 + w) * 16;

  // A operands for all 4 K-steps: agg tile and h tile (16 x 128 bf16 each)
  v16bf aA[4], aX[4];
  {
    const __bf16* abase = aggb + (size_t)(node0 + row) * DIM + half * 8;
    const __bf16* xbase = xb   + (size_t)(node0 + row) * DIM + half * 8;
#pragma unroll
    for (int ks = 0; ks < 4; ++ks) {
      BF16x16 ta, tx;
      ta.q[0] = *(const u32x4*)(abase + ks * 32);
      ta.q[1] = *(const u32x4*)(abase + ks * 32 + 16);
      tx.q[0] = *(const u32x4*)(xbase + ks * 32);
      tx.q[1] = *(const u32x4*)(xbase + ks * 32 + 16);
      aA[ks] = ta.v;
      aX[ks] = tx.v;
    }
  }

  v8f zero = {};
#pragma unroll
  for (int ct = 0; ct < 8; ++ct) {
    v8f gi[3], gh[3];
#pragma unroll
    for (int g = 0; g < 3; ++g) { gi[g] = zero; gh[g] = zero; }

#pragma unroll
    for (int ks = 0; ks < 4; ++ks) {
      const int kk = ks * 32 + half * 8;     // B operand K base (table order), N = row
      const __bf16* bi = wihT + (size_t)kk * 384 + row;
      const __bf16* bh = whhT + (size_t)kk * 384 + row;
#pragma unroll
      for (int g = 0; g < 3; ++g) {
        const int j = g * 128 + ct * 16;
        BF16x16 b;
#pragma unroll
        for (int s = 0; s < 16; ++s) b.v[s] = bi[(s < 8 ? s : s + 8) * 384 + j];
        gi[g] = wmma_bf16(aA[ks], b.v, gi[g]);
#pragma unroll
        for (int s = 0; s < 16; ++s) b.v[s] = bh[(s < 8 ? s : s + 8) * 384 + j];
        gh[g] = wmma_bf16(aX[ks], b.v, gh[g]);
      }
    }

    const int c = ct * 16 + row;
    const float bir = b_ih[c], biz = b_ih[128 + c], bin = b_ih[256 + c];
    const float bhr = b_hh[c], bhz = b_hh[128 + c], bhn = b_hh[256 + c];
#pragma unroll
    for (int v = 0; v < 8; ++v) {
      const int node = node0 + half * 8 + v;
      float ir = gi[0][v] + bir, iz = gi[1][v] + biz, inn = gi[2][v] + bin;
      float hr = gh[0][v] + bhr, hz = gh[1][v] + bhz, hn  = gh[2][v] + bhn;
      float r = sigmoidf_(ir + hr);
      float z = sigmoidf_(iz + hz);
      float n = tanhf(inn + r * hn);
      float x = X[(size_t)node * DIM + c];
      out[(size_t)node * DIM + c] = (1.0f - z) * n + z * x;
    }
  }
}

// ---------------------------------------------------------------------------
// Launch
// ---------------------------------------------------------------------------
extern "C" void kernel_launch(void* const* d_in, const int* in_sizes, int n_in,
                              void* d_out, int out_size, void* d_ws, size_t ws_size,
                              hipStream_t stream) {
  const float* X   = (const float*)d_in[0];   // node_features (8192x128)
  const int*   A   = (const int*)d_in[1];     // adj_matrix    (8192x8192)
  const float* Wm  = (const float*)d_in[2];   // w_msg (128x128)
  const float* bm  = (const float*)d_in[3];   // b_msg (128)
  const float* Wih = (const float*)d_in[4];   // w_ih (384x128)
  const float* Whh = (const float*)d_in[5];   // w_hh (384x128)
  const float* bih = (const float*)d_in[6];   // b_ih (384)
  const float* bhh = (const float*)d_in[7];   // b_hh (384)
  float* out = (float*)d_out;

  char* ws = (char*)d_ws;
  const size_t MB2 = (size_t)NNODES * DIM * sizeof(__bf16);   // 2 MiB
  __bf16* msgb = (__bf16*)(ws);
  __bf16* xb   = (__bf16*)(ws + MB2);
  __bf16* aggb = (__bf16*)(ws + 2 * MB2);
  __bf16* wihT = (__bf16*)(ws + 3 * MB2);
  __bf16* whhT = (__bf16*)(ws + 3 * MB2 + 384 * 128 * sizeof(__bf16));

  k_prep<<<dim3(192), dim3(256), 0, stream>>>(Wih, Whh, wihT, whhT);
  k_msg <<<dim3(NNODES / 16), dim3(256), 0, stream>>>(X, Wm, bm, msgb, xb);
  k_agg <<<dim3(NNODES / 64), dim3(128), 0, stream>>>(A, msgb, aggb);
  k_gru <<<dim3(NNODES / 64), dim3(128), 0, stream>>>(aggb, xb, X, wihT, whhT,
                                                      bih, bhh, out);
}
